// MultiHeadAttention_5609227288661
// MI455X (gfx1250) — matrix operand order repaired and checked by reading, not compiled, not run on hardware
//
#include <hip/hip_runtime.h>

#define HH 8
#define DK 64
#define DM 512
#define MAXREL 512
#define NEGINF (-1000000000.0f)

typedef __attribute__((ext_vector_type(16))) _Float16 v16h;
typedef __attribute__((ext_vector_type(8)))  float    v8f;
typedef __attribute__((ext_vector_type(4)))  unsigned int u32x4;
typedef __attribute__((ext_vector_type(8)))  int      i32x8;
typedef __attribute__((ext_vector_type(4)))  int      i32x4;

union Frag { v16h v; uint4 u[2]; };

__device__ __forceinline__ v8f wmma16(v16h a, v16h b, v8f c) {
  // D = A(16x32,f16) * B(32x16,f16) + C(16x16,f32)
  return __builtin_amdgcn_wmma_f32_16x16x32_f16(false, a, false, b, (short)0, c,
                                                false, false);
}

__device__ __forceinline__ float rmax16(float x) {
  x = fmaxf(x, __shfl_xor(x, 1));
  x = fmaxf(x, __shfl_xor(x, 2));
  x = fmaxf(x, __shfl_xor(x, 4));
  x = fmaxf(x, __shfl_xor(x, 8));
  return x;
}
__device__ __forceinline__ float rsum16(float x) {
  x += __shfl_xor(x, 1);
  x += __shfl_xor(x, 2);
  x += __shfl_xor(x, 4);
  x += __shfl_xor(x, 8);
  return x;
}

// ---------------------------------------------------------------------------
// TDM: DMA a 64-wide (halves) x `rows` 2-D tile from a row-major [*,64] f16
// tensor into LDS, inserting 4 dwords of padding after every 32 dwords so the
// LDS image has a 72-half row stride (matches conflict-free fragment reads).
// D# layout per CDNA5 ISA ch.8 (group0 128b, group1 256b; groups 2/3 unused).
// ---------------------------------------------------------------------------
__device__ __forceinline__ void tdm_load_tile64(const _Float16* gsrc,
                                                _Float16* ldst, int rows) {
  const unsigned lds_off = (unsigned)(size_t)ldst;  // flat addr low 32 = LDS offset
  const unsigned long long ga = (unsigned long long)(size_t)gsrc;
  u32x4 g0;
  g0[0] = 1u;                                        // count=1 (valid user D#)
  g0[1] = lds_off;                                   // lds_addr (bytes)
  g0[2] = (unsigned)(ga & 0xFFFFFFFFull);            // global_addr[31:0]
  g0[3] = (unsigned)((ga >> 32) & 0x01FFFFFFull) | (2u << 30);  // addr[56:32]|type=2
  i32x8 g1;
  g1[0] = (1 << 16)      // data_size = 1 -> 2 bytes
        | (1 << 20)      // pad_enable
        | (4 << 22)      // pad_interval code 4 -> every 32 dwords
        | (3 << 25);     // pad_amount  code 3 -> 4 dwords (8 halves)
  g1[1] = (64 << 16);    // tensor_dim0 = 64 elements
  g1[2] = (rows << 16);  // tensor_dim1
  g1[3] = (64 << 16);    // tile_dim0 = 64
  g1[4] = rows;          // tile_dim1
  g1[5] = 64;            // tensor_dim0_stride = 64 elements
  g1[6] = 0;
  g1[7] = 0;
  const i32x4 z4 = {0, 0, 0, 0};
#if __clang_major__ >= 23
  const i32x8 z8 = {0, 0, 0, 0, 0, 0, 0, 0};
  __builtin_amdgcn_tensor_load_to_lds(g0, g1, z4, z4, z8, 0);
#else
  __builtin_amdgcn_tensor_load_to_lds(g0, g1, z4, z4, 0);
#endif
}

// ---------------------------------------------------------------------------
// Tiled WMMA GEMM:  out[m,n] = sum_k A[m,k] * W[n,k] + bias[n]
// Block = 256 threads (8 waves), tile 128(M) x 64(N), k-step 32 via LDS (f16).
// ---------------------------------------------------------------------------
template <typename TIN>
__global__ void __launch_bounds__(256)
gemm_wmma_kernel(const TIN* __restrict__ A, const float* __restrict__ W,
                 const float* __restrict__ bias,
                 _Float16* __restrict__ outH, float* __restrict__ outF,
                 int N, int K,
                 long strideA, long strideW, int wmod, long strideOH,
                 int headsplit, int Ldim) {
  __shared__ __align__(16) _Float16 sA[128 * 40];  // row stride 40 halves
  __shared__ __align__(16) _Float16 sB[64 * 40];

  const int t = threadIdx.x, wave = t >> 5, lane = t & 31;
  const int z = blockIdx.z;
  A += (long)z * strideA;
  W += (long)(z % wmod) * strideW;
  if (outH) outH += (long)z * strideOH;
  const int m0 = blockIdx.y * 128, n0 = blockIdx.x * 64;

  v8f acc[4];
#pragma unroll
  for (int nt = 0; nt < 4; ++nt)
#pragma unroll
    for (int e = 0; e < 8; ++e) acc[nt][e] = 0.f;

  const int arow = t >> 1, ach = (t & 1) * 16;  // 256 units of 16 elems (A)
  const int lanelo = lane & 15, hv = lane >> 4;

  for (int k0 = 0; k0 < K; k0 += 32) {
    {  // stage A tile 128x32 -> f16 LDS
      const TIN* src = A + (long)(m0 + arow) * K + k0 + ach;
      _Float16* dst = &sA[arow * 40 + ach];
#pragma unroll
      for (int u = 0; u < 16; ++u) dst[u] = (_Float16)(float)src[u];
      if (k0 + 32 < K) __builtin_prefetch(src + 32, 0, 0);
    }
    if (t < 128) {  // stage B tile 64x32 (rows of W) -> f16 LDS
      const float* src = W + (long)(n0 + arow) * K + k0 + ach;
      _Float16* dst = &sB[arow * 40 + ach];
#pragma unroll
      for (int u = 0; u < 16; ++u) dst[u] = (_Float16)src[u];
      if (k0 + 32 < K) __builtin_prefetch(src + 32, 0, 0);
    }
    __syncthreads();

    Frag a;  // A frag: lane<16 -> M=lane, K{0-7,16-23}; lane>=16 -> K{8-15,24-31}
    {
      const int ar = wave * 16 + lanelo;
      const int kb = hv ? 8 : 0;
      a.u[0] = *(const uint4*)&sA[ar * 40 + kb];
      a.u[1] = *(const uint4*)&sA[ar * 40 + kb + 16];
    }
#pragma unroll
    for (int nt = 0; nt < 4; ++nt) {
      Frag b;  // B frag: lane<16 -> K{0-7,16-23}; lane>=16 -> K{8-15,24-31} at col n
      const int br = nt * 16 + lanelo;
      const int db = hv ? 8 : 0;
      b.u[0] = *(const uint4*)&sB[br * 40 + db];
      b.u[1] = *(const uint4*)&sB[br * 40 + db + 16];
      acc[nt] = wmma16(a.v, b.v, acc[nt]);
    }
    __syncthreads();
  }

  // epilogue: C layout M = e + 8*hv, N = lanelo
  const int mloc = wave * 16 + 8 * hv;
#pragma unroll
  for (int nt = 0; nt < 4; ++nt) {
    const int col = n0 + nt * 16 + lanelo;
    const float bval = bias ? bias[col] : 0.f;
#pragma unroll
    for (int e = 0; e < 8; ++e) {
      const int row = m0 + mloc + e;
      const float val = acc[nt][e] + bval;
      if (outF) outF[(long)row * N + col] = val;
      if (outH) {
        long idx;
        if (headsplit) {
          const int bb = row / Ldim, ll = row - bb * Ldim;
          idx = (((long)(bb * HH + (col >> 6)) * Ldim + ll) * DK + (col & 63));
        } else {
          idx = (long)row * N + col;
        }
        outH[idx] = (_Float16)val;
      }
    }
  }
}

// ---------------------------------------------------------------------------
// Flash attention, causal + relative-position bias, 64-key steps.
// Block = 128 threads (4 waves); block owns 64 query rows, wave w owns 16.
// K tile DMA'd by the Tensor Data Mover (wave 0) with TDM-generated padding;
// V tile staged transposed by all threads. S = 4 tiles x 2 WMMAs; online
// softmax with shfl reductions; P (16x64) -> LDS -> 2 A frags; O += P*V via
// 8 WMMAs.
// ---------------------------------------------------------------------------
__global__ void __launch_bounds__(128)
attn_kernel(const _Float16* __restrict__ Qh, const _Float16* __restrict__ Kh,
            const _Float16* __restrict__ Vh, const _Float16* __restrict__ Rh,
            _Float16* __restrict__ Oc, int L) {
  __shared__ __align__(16) _Float16 sK[64 * 72];      // 64 keys x 64 d (stride 72)
  __shared__ __align__(16) _Float16 sV[64 * 72];      // transposed: 64 d x 64 keys
  __shared__ __align__(16) _Float16 sP[4 * 16 * 72];  // per-wave P 16x64

  const int t = threadIdx.x, wave = t >> 5, lane = t & 31;
  const int i0 = blockIdx.x * 64, h = blockIdx.y, b = blockIdx.z;
  const long bh = (long)(b * HH + h);
  const _Float16* Q = Qh + bh * (long)L * DK;
  const _Float16* Kp = Kh + bh * (long)L * DK;
  const _Float16* Vp = Vh + bh * (long)L * DK;
  const _Float16* R = Rh + bh * (long)L * DM;

  const int lanelo = lane & 15, hv = lane >> 4;
  const int iw = i0 + wave * 16;

  Frag qa[2];  // Q rows iw..iw+15, K(=d) 0..31 and 32..63
  {
    const _Float16* qr = Q + (long)(iw + lanelo) * DK;
    const int kb = hv ? 8 : 0;
    qa[0].u[0] = *(const uint4*)&qr[kb];
    qa[0].u[1] = *(const uint4*)&qr[kb + 16];
    qa[1].u[0] = *(const uint4*)&qr[32 + kb];
    qa[1].u[1] = *(const uint4*)&qr[32 + kb + 16];
  }

  v8f o[4];
  float mrow[8], lrow[8];
#pragma unroll
  for (int dt = 0; dt < 4; ++dt)
#pragma unroll
    for (int e = 0; e < 8; ++e) o[dt][e] = 0.f;
#pragma unroll
  for (int e = 0; e < 8; ++e) { mrow[e] = -3.0e38f; lrow[e] = 0.f; }

  const int nsteps = i0 / 64 + 1;               // keys 0 .. i0+63
  const int srow = t >> 1, sch = (t & 1) * 32;  // V staging: 64 rows x 2 chunks
  _Float16* myP = &sP[wave * 16 * 72];

  for (int s = 0; s < nsteps; ++s) {
    const int j0 = s * 64;
    if (wave == 0) {  // TDM DMA of K tile (64x64 halves) into padded LDS image
      tdm_load_tile64(Kp + (long)j0 * DK, sK, 64);
    }
    {  // stage V tile transposed: sV[d][j]
      const _Float16* vs = Vp + (long)(j0 + srow) * DK + sch;
      uint4 vv[4];
      vv[0] = *(const uint4*)&vs[0];
      vv[1] = *(const uint4*)&vs[8];
      vv[2] = *(const uint4*)&vs[16];
      vv[3] = *(const uint4*)&vs[24];
      const _Float16* ve = (const _Float16*)&vv[0];
#pragma unroll
      for (int u = 0; u < 32; ++u) sV[(sch + u) * 72 + srow] = ve[u];
    }
    if (wave == 0) __builtin_amdgcn_s_wait_tensorcnt(0);
    __syncthreads();

    if (j0 <= iw + 15) {  // any unmasked column for this wave?
      v8f S[4];
#pragma unroll
      for (int jt = 0; jt < 4; ++jt) {
        Frag b0, b1;
        const int kr = (jt * 16 + lanelo) * 72;
        const int db = hv ? 8 : 0;
        b0.u[0] = *(const uint4*)&sK[kr + db];
        b0.u[1] = *(const uint4*)&sK[kr + db + 16];
        b1.u[0] = *(const uint4*)&sK[kr + 32 + db];
        b1.u[1] = *(const uint4*)&sK[kr + 32 + db + 16];
        v8f c;
#pragma unroll
        for (int e = 0; e < 8; ++e) c[e] = 0.f;
        c = wmma16(qa[0].v, b0.v, c);
        c = wmma16(qa[1].v, b1.v, c);
        S[jt] = c;
      }
      // causal mask + relative-position bias (band width 512)
      const int ib = iw + 8 * hv;
#pragma unroll
      for (int jt = 0; jt < 4; ++jt) {
#pragma unroll
        for (int e = 0; e < 8; ++e) {
          const int i = ib + e;
          const int j = j0 + jt * 16 + lanelo;
          float sv = S[jt][e];
          if (j > i) {
            sv = NEGINF;
          } else {
            const int r = i - j;
            if (r < MAXREL) sv += (float)R[(long)i * DM + (MAXREL - 1 - r)];
          }
          S[jt][e] = sv;
        }
      }
      // online softmax (row = e + 8*hv lives across one 16-lane half)
#pragma unroll
      for (int e = 0; e < 8; ++e) {
        const float tm = rmax16(fmaxf(fmaxf(S[0][e], S[1][e]),
                                      fmaxf(S[2][e], S[3][e])));
        const float mn = fmaxf(mrow[e], tm);
        const float alpha = __expf(mrow[e] - mn);
        mrow[e] = mn;
        float ps = 0.f;
#pragma unroll
        for (int jt = 0; jt < 4; ++jt) {
          const float p = __expf(S[jt][e] - mn);
          S[jt][e] = p;
          ps += p;
        }
        lrow[e] = lrow[e] * alpha + rsum16(ps);
#pragma unroll
        for (int dt = 0; dt < 4; ++dt) o[dt][e] *= alpha;
      }
      // P: C layout -> LDS (16x64 f16) -> A fragments (wave-private)
#pragma unroll
      for (int jt = 0; jt < 4; ++jt)
#pragma unroll
        for (int e = 0; e < 8; ++e)
          myP[(e + 8 * hv) * 72 + jt * 16 + lanelo] = (_Float16)S[jt][e];
      Frag pa0, pa1;
      {
        const int kb = hv ? 8 : 0;
        pa0.u[0] = *(const uint4*)&myP[lanelo * 72 + kb];
        pa0.u[1] = *(const uint4*)&myP[lanelo * 72 + kb + 16];
        pa1.u[0] = *(const uint4*)&myP[lanelo * 72 + 32 + kb];
        pa1.u[1] = *(const uint4*)&myP[lanelo * 72 + 32 + kb + 16];
      }
#pragma unroll
      for (int dt = 0; dt < 4; ++dt) {
        Frag vb0, vb1;
        const int dr = (dt * 16 + lanelo) * 72;
        const int jb = hv ? 8 : 0;
        vb0.u[0] = *(const uint4*)&sV[dr + jb];
        vb0.u[1] = *(const uint4*)&sV[dr + jb + 16];
        vb1.u[0] = *(const uint4*)&sV[dr + 32 + jb];
        vb1.u[1] = *(const uint4*)&sV[dr + 32 + jb + 16];
        o[dt] = wmma16(pa0.v, vb0.v, o[dt]);
        o[dt] = wmma16(pa1.v, vb1.v, o[dt]);
      }
    }
    __syncthreads();
  }

  // normalize + write concat layout [B, L, H*DK]
#pragma unroll
  for (int e = 0; e < 8; ++e) {
    const float inv = 1.f / lrow[e];
    const int i = iw + e + 8 * hv;
#pragma unroll
    for (int dt = 0; dt < 4; ++dt) {
      const int d = dt * 16 + lanelo;
      Oc[((long)b * L + i) * DM + h * DK + d] = (_Float16)(o[dt][e] * inv);
    }
  }
}

// ---------------------------------------------------------------------------
extern "C" void kernel_launch(void* const* d_in, const int* in_sizes, int n_in,
                              void* d_out, int out_size, void* d_ws,
                              size_t ws_size, hipStream_t stream) {
  (void)in_sizes; (void)n_in; (void)out_size; (void)ws_size;
  const float* q  = (const float*)d_in[0];
  const float* k  = (const float*)d_in[1];
  const float* v  = (const float*)d_in[2];
  // d_in[3] = mask (causal, implicit)
  const float* Wq = (const float*)d_in[4];
  const float* bq = (const float*)d_in[5];
  const float* Wk = (const float*)d_in[6];
  const float* bk = (const float*)d_in[7];
  const float* Wv = (const float*)d_in[8];
  const float* bv = (const float*)d_in[9];
  const float* Wo = (const float*)d_in[10];
  const float* bo = (const float*)d_in[11];
  const float* rel = (const float*)d_in[12];

  const int B = 2, L = 2048;
  const long nqkv = (long)B * HH * L * DK;  // == B*L*DM elements
  _Float16* Qh = (_Float16*)d_ws;
  _Float16* Kh = Qh + nqkv;
  _Float16* Vh = Kh + nqkv;
  _Float16* Oc = Vh + nqkv;
  _Float16* Rh = Oc + nqkv;  // [B,H,L,512] f16 rel logits
  float* out = (float*)d_out;

  const dim3 blk(256);
  const dim3 gproj(DM / 64, (B * L) / 128, 1);

  // Q/K/V projections -> f16 head-split [B,H,L,64]
  gemm_wmma_kernel<float><<<gproj, blk, 0, stream>>>(
      q, Wq, bq, Qh, nullptr, DM, DM, 0, 0, 1, 0, 1, L);
  gemm_wmma_kernel<float><<<gproj, blk, 0, stream>>>(
      k, Wk, bk, Kh, nullptr, DM, DM, 0, 0, 1, 0, 1, L);
  gemm_wmma_kernel<float><<<gproj, blk, 0, stream>>>(
      v, Wv, bv, Vh, nullptr, DM, DM, 0, 0, 1, 0, 1, L);

  // R[b,h] = Qh[b,h] (L x 64) @ rel_emb[h]^T (512 x 64)^T  -> f16 [L,512]
  const dim3 grel(DM / 64, L / 128, B * HH);
  gemm_wmma_kernel<_Float16><<<grel, blk, 0, stream>>>(
      Qh, rel, nullptr, Rh, nullptr, DM, DK,
      (long)L * DK, (long)MAXREL * DK, HH, (long)L * DM, 0, L);

  // flash attention -> concat f16 [B, L, 512]
  attn_kernel<<<dim3(L / 64, HH, B), dim3(128), 0, stream>>>(
      Qh, Kh, Vh, Rh, Oc, L);

  // output projection -> f32 d_out
  gemm_wmma_kernel<_Float16><<<gproj, blk, 0, stream>>>(
      Oc, Wo, bo, nullptr, out, DM, DM, 0, 0, 1, 0, 0, L);
}
